// GraphVAE_2482491097961
// MI455X (gfx1250) — matrix operand order repaired and checked by reading, not compiled, not run on hardware
//
#include <hip/hip_runtime.h>
#include <hip/hip_bf16.h>

typedef __attribute__((ext_vector_type(16))) _Float16 v16h;
typedef __attribute__((ext_vector_type(8)))  _Float16 v8h;
typedef __attribute__((ext_vector_type(8)))  float    v8f;

#define GN      50000
#define GE      800000
#define IN_CH   128
#define HID     128
#define LAT     64
#define NGRAPH  256
#define MAXN    320
#define GSCALE  0.125f   // 64^-0.5

__device__ __forceinline__ v16h cat8(v8h lo, v8h hi) {
  return __builtin_shufflevector(lo, hi, 0,1,2,3,4,5,6,7,8,9,10,11,12,13,14,15);
}

// ---------------- prep kernels ----------------

// transpose + fp16-convert weights: Wt[n][k] = (f16)W[k][n]; Wml = [Wmu|Wlv]
__global__ void k_prep_weights(const float* __restrict__ W1, const float* __restrict__ W2,
                               const float* __restrict__ Wmu, const float* __restrict__ Wlv,
                               _Float16* __restrict__ Wt1, _Float16* __restrict__ Wt2,
                               _Float16* __restrict__ Wml) {
  int t = blockIdx.x * blockDim.x + threadIdx.x;
  if (t >= 128 * 128) return;
  int k = t >> 7, n = t & 127;
  Wt1[n * 128 + k] = (_Float16)W1[k * 128 + n];
  Wt2[n * 128 + k] = (_Float16)W2[k * 128 + n];
  _Float16 wv = (n < 64) ? (_Float16)Wmu[k * 64 + n] : (_Float16)Wlv[k * 64 + (n - 64)];
  Wml[n * 128 + k] = wv;
}

__global__ void k_f32_to_f16(const float* __restrict__ in, _Float16* __restrict__ out, int n) {
  int t = blockIdx.x * blockDim.x + threadIdx.x;
  if (t < n) out[t] = (_Float16)in[t];
}

__global__ void k_fill_f32(float* __restrict__ p, float v, int n) {
  int t = blockIdx.x * blockDim.x + threadIdx.x;
  if (t < n) p[t] = v;
}

__global__ void k_fill_u32(unsigned int* __restrict__ p, unsigned int v, int n) {
  int t = blockIdx.x * blockDim.x + threadIdx.x;
  if (t < n) p[t] = v;
}

// ---------------- graph-structure kernels ----------------

__global__ void k_deg_edges(const int* __restrict__ dst, float* __restrict__ deg, int e) {
  int t = blockIdx.x * blockDim.x + threadIdx.x;
  if (t < e) atomicAdd(&deg[dst[t]], 1.0f);
}

__global__ void k_dinv(float* __restrict__ deg, int n) {
  int t = blockIdx.x * blockDim.x + threadIdx.x;
  if (t < n) { float d = deg[t]; deg[t] = (d > 0.f) ? rsqrtf(d) : 0.f; }
}

__global__ void k_firstocc(const int* __restrict__ batch, int* __restrict__ fo, int n) {
  int t = blockIdx.x * blockDim.x + threadIdx.x;
  if (t < n) atomicMin(&fo[batch[t]], t);
}

// ---------------- WMMA GEMM: C[rows x 128] = A16[rows x 128] @ Wt^T ----------------
// Wt stored [n][k] (128x128 f16). One wave per 16-row strip, 8 col tiles, K=128.
__global__ void k_gemm_f16(const _Float16* __restrict__ A, const _Float16* __restrict__ Wt,
                           float* __restrict__ C, int rows) {
  int wave = blockIdx.x * (blockDim.x >> 5) + (threadIdx.x >> 5);
  int m0 = wave * 16;
  if (m0 >= rows) return;              // wave-uniform -> EXEC stays all-1s for WMMA
  int lane = threadIdx.x & 31;
  int lm = lane & 15, lh = lane >> 4;

  const _Float16* arow = A + (size_t)(m0 + lm) * 128;
  v16h a[4];
#pragma unroll
  for (int kc = 0; kc < 4; ++kc) {
    int kb = kc * 32 + lh * 8;
    v8h lo = *(const v8h*)(arow + kb);
    v8h hi = *(const v8h*)(arow + kb + 16);
    a[kc] = cat8(lo, hi);
  }
#pragma unroll
  for (int nt = 0; nt < 8; ++nt) {
    int n0 = nt * 16;
    const _Float16* wrow = Wt + (size_t)(n0 + lm) * 128;
    v8f c = {};
#pragma unroll
    for (int kc = 0; kc < 4; ++kc) {
      int kb = kc * 32 + lh * 8;
      v8h lo = *(const v8h*)(wrow + kb);
      v8h hi = *(const v8h*)(wrow + kb + 16);
      v16h b = cat8(lo, hi);
      c = __builtin_amdgcn_wmma_f32_16x16x32_f16(false, a[kc], false, b,
                                                 (short)0, c, false, false);
    }
    float* cout = C + (size_t)(m0 + (lh << 3)) * 128 + n0 + lm;
#pragma unroll
    for (int v = 0; v < 8; ++v) cout[(size_t)v * 128] = c[v];
  }
}

// ---------------- aggregation ----------------

// self-loop contribution: acc = dinv[i]^2 * lin
__global__ void k_acc_init(const float* __restrict__ dinv, const float* __restrict__ lin,
                           float* __restrict__ acc, int n128) {
  int t = blockIdx.x * blockDim.x + threadIdx.x;
  if (t >= n128) return;
  float di = dinv[t >> 7];
  acc[t] = di * di * lin[t];
}

// 32 threads/edge, float4 each -> acc[dst] += w * lin[src]
__global__ void k_scatter(const int* __restrict__ src, const int* __restrict__ dst,
                          const float* __restrict__ dinv, const float* __restrict__ lin,
                          float* __restrict__ acc, int e) {
  long long t = (long long)blockIdx.x * blockDim.x + threadIdx.x;
  if (t >= (long long)e * 32) return;
  int ed = (int)(t >> 5);
  int f = ((int)t & 31) * 4;
  int s = src[ed], d = dst[ed];
  float w = dinv[s] * dinv[d];
  const float4 v = *(const float4*)(lin + (size_t)s * 128 + f);
  float* o = acc + (size_t)d * 128 + f;
  atomicAdd(o + 0, w * v.x);
  atomicAdd(o + 1, w * v.y);
  atomicAdd(o + 2, w * v.z);
  atomicAdd(o + 3, w * v.w);
}

__global__ void k_bias_relu_f16(const float* __restrict__ acc, const float* __restrict__ b,
                                _Float16* __restrict__ out, int n128) {
  int t = blockIdx.x * blockDim.x + threadIdx.x;
  if (t >= n128) return;
  float v = acc[t] + b[t & 127];
  out[t] = (_Float16)fmaxf(v, 0.f);
}

// ---------------- VAE head: mu/logvar/z + dense scatter ----------------
__global__ void k_mulv(const float* __restrict__ lin, const float* __restrict__ bmu,
                       const float* __restrict__ blv, const float* __restrict__ eps,
                       const int* __restrict__ batch, const int* __restrict__ fo,
                       float* __restrict__ mu_o, float* __restrict__ lv_o,
                       float* __restrict__ mask_o, _Float16* __restrict__ Zd, int n) {
  int t = blockIdx.x * blockDim.x + threadIdx.x;
  if (t >= n * LAT) return;
  int i = t >> 6, f = t & 63;
  float mu = lin[(size_t)i * 128 + f] + bmu[f];
  float lv = lin[(size_t)i * 128 + 64 + f] + blv[f];
  mu_o[t] = mu;
  lv_o[t] = lv;
  float lvc = fminf(fmaxf(lv, -20.f), 20.f);
  float z = mu + eps[t] * expf(0.5f * lvc);
  int b = batch[i];
  int pos = i - fo[b];
  if (pos >= 0 && pos < MAXN) {
    Zd[((size_t)b * MAXN + pos) * LAT + f] = (_Float16)z;
    if (f == 0) mask_o[b * MAXN + pos] = 1.0f;
  }
}

// ---------------- decoder: adj = sigmoid(scale * Z Z^T + bias), WMMA f16 ----------------
// one wave per 16x16 output tile; 256 graphs * 20*20 tiles = 102400 waves (exact grid)
__global__ void k_adj(const _Float16* __restrict__ Z, const float* __restrict__ biasp,
                      float* __restrict__ adj) {
  int wave = blockIdx.x * (blockDim.x >> 5) + (threadIdx.x >> 5);
  int g = wave / 400;
  int t = wave % 400;
  int m0 = (t / 20) * 16, n0 = (t % 20) * 16;
  int lane = threadIdx.x & 31;
  int lm = lane & 15, lh = lane >> 4;
  const _Float16* Zg = Z + (size_t)g * MAXN * LAT;
  const _Float16* arow = Zg + (size_t)(m0 + lm) * LAT;
  const _Float16* brow = Zg + (size_t)(n0 + lm) * LAT;
  v8f c = {};
#pragma unroll
  for (int kc = 0; kc < 2; ++kc) {
    int k0 = kc * 32;
    v8h alo = *(const v8h*)(arow + k0 + lh * 8);
    v8h ahi = *(const v8h*)(arow + k0 + lh * 8 + 16);
    v8h blo = *(const v8h*)(brow + k0 + lh * 8);
    v8h bhi = *(const v8h*)(brow + k0 + lh * 8 + 16);
    c = __builtin_amdgcn_wmma_f32_16x16x32_f16(false, cat8(alo, ahi), false, cat8(blo, bhi),
                                               (short)0, c, false, false);
  }
  float bias = biasp[0];
  float* out = adj + (size_t)g * MAXN * MAXN + (size_t)(m0 + (lh << 3)) * MAXN + n0 + lm;
#pragma unroll
  for (int v = 0; v < 8; ++v) {
    float s = GSCALE * c[v] + bias;
    out[(size_t)v * MAXN] = 1.0f / (1.0f + expf(-s));
  }
}

// ---------------- host launcher ----------------

static inline char* bump(char*& p, size_t bytes) {
  char* r = p;
  p += (bytes + 255) & ~(size_t)255;
  return r;
}

extern "C" void kernel_launch(void* const* d_in, const int* in_sizes, int n_in,
                              void* d_out, int out_size, void* d_ws, size_t ws_size,
                              hipStream_t stream) {
  const float* x    = (const float*)d_in[0];
  const int*   ei   = (const int*)d_in[1];     // (2,E): src = ei, dst = ei+E
  const int*   batch= (const int*)d_in[2];
  const float* eps  = (const float*)d_in[3];
  const float* W1   = (const float*)d_in[4];
  const float* b1   = (const float*)d_in[5];
  const float* W2   = (const float*)d_in[6];
  const float* b2   = (const float*)d_in[7];
  const float* Wmu  = (const float*)d_in[8];
  const float* bmu  = (const float*)d_in[9];
  const float* Wlv  = (const float*)d_in[10];
  const float* blv  = (const float*)d_in[11];
  const float* dbias= (const float*)d_in[12];

  const int* src = ei;
  const int* dst = ei + GE;

  // outputs: adj | mu | logvar | mask
  float* adj_o  = (float*)d_out;
  float* mu_o   = adj_o + (size_t)NGRAPH * MAXN * MAXN;
  float* lv_o   = mu_o + (size_t)GN * LAT;
  float* mask_o = lv_o + (size_t)GN * LAT;

  // workspace
  char* p = (char*)d_ws;
  float*     dinv = (float*)    bump(p, (size_t)GN * 4);
  _Float16*  Wt1  = (_Float16*) bump(p, 128 * 128 * 2);
  _Float16*  Wt2  = (_Float16*) bump(p, 128 * 128 * 2);
  _Float16*  Wml  = (_Float16*) bump(p, 128 * 128 * 2);
  _Float16*  A16  = (_Float16*) bump(p, (size_t)GN * 128 * 2);
  float*     lin  = (float*)    bump(p, (size_t)GN * 128 * 4);
  float*     acc  = (float*)    bump(p, (size_t)GN * 128 * 4);
  int*       fo   = (int*)      bump(p, NGRAPH * 4);
  _Float16*  Zd   = (_Float16*) bump(p, (size_t)NGRAPH * MAXN * LAT * 2);

  const int B = 256;
  const int n128 = GN * 128;          // 6,400,000
  const int nlat = GN * LAT;          // 3,200,000
  const int zdw  = NGRAPH * MAXN * LAT / 2;   // Zd as u32 words

  // prep
  k_prep_weights<<<(128 * 128 + B - 1) / B, B, 0, stream>>>(W1, W2, Wmu, Wlv, Wt1, Wt2, Wml);
  k_f32_to_f16<<<(n128 + B - 1) / B, B, 0, stream>>>(x, A16, n128);
  k_fill_f32<<<(GN + B - 1) / B, B, 0, stream>>>(dinv, 1.0f, GN);          // deg incl. self-loop
  k_fill_u32<<<(NGRAPH + B - 1) / B, B, 0, stream>>>((unsigned int*)fo, 0x7FFFFFFFu, NGRAPH);
  k_fill_u32<<<(zdw + B - 1) / B, B, 0, stream>>>((unsigned int*)Zd, 0u, zdw);
  k_fill_f32<<<(NGRAPH * MAXN + B - 1) / B, B, 0, stream>>>(mask_o, 0.0f, NGRAPH * MAXN);

  // degree -> dinv
  k_deg_edges<<<(GE + B - 1) / B, B, 0, stream>>>(dst, dinv, GE);
  k_dinv<<<(GN + B - 1) / B, B, 0, stream>>>(dinv, GN);

  const int gemm_blocks = ((GN + 15) / 16 + 7) / 8;   // 8 waves/block, 16 rows/wave
  const long long scat_total = (long long)GE * 32;
  const int scat_blocks = (int)((scat_total + B - 1) / B);

  // GCN layer 1
  k_gemm_f16<<<gemm_blocks, B, 0, stream>>>(A16, Wt1, lin, GN);
  k_acc_init<<<(n128 + B - 1) / B, B, 0, stream>>>(dinv, lin, acc, n128);
  k_scatter<<<scat_blocks, B, 0, stream>>>(src, dst, dinv, lin, acc, GE);
  k_bias_relu_f16<<<(n128 + B - 1) / B, B, 0, stream>>>(acc, b1, A16, n128);

  // GCN layer 2
  k_gemm_f16<<<gemm_blocks, B, 0, stream>>>(A16, Wt2, lin, GN);
  k_acc_init<<<(n128 + B - 1) / B, B, 0, stream>>>(dinv, lin, acc, n128);
  k_scatter<<<scat_blocks, B, 0, stream>>>(src, dst, dinv, lin, acc, GE);
  k_bias_relu_f16<<<(n128 + B - 1) / B, B, 0, stream>>>(acc, b2, A16, n128);

  // heads (mu | logvar fused)
  k_gemm_f16<<<gemm_blocks, B, 0, stream>>>(A16, Wml, lin, GN);

  // dense batch positions + reparameterize + scatter z
  k_firstocc<<<(GN + B - 1) / B, B, 0, stream>>>(batch, fo, GN);
  k_mulv<<<(nlat + B - 1) / B, B, 0, stream>>>(lin, bmu, blv, eps, batch, fo,
                                               mu_o, lv_o, mask_o, Zd, GN);

  // decoder: 256 graphs * 400 tiles, 8 waves/block -> 12800 blocks exactly
  k_adj<<<NGRAPH * 400 / 8, B, 0, stream>>>(Zd, dbias, adj_o);
}
